// GAT_68255620268395
// MI455X (gfx1250) — matrix operand order repaired and checked by reading, not compiled, not run on hardware
//
#include <hip/hip_runtime.h>
#include <hip/hip_bf16.h>
#include <math.h>

typedef __attribute__((ext_vector_type(16))) _Float16     v16h;
typedef __attribute__((ext_vector_type(8)))  float        v8f;
typedef __attribute__((ext_vector_type(4)))  unsigned int u32x4;
typedef __attribute__((ext_vector_type(8)))  int          i32x8;
typedef __attribute__((ext_vector_type(4)))  int          i32x4;

#define HEADS 8
#define CH    8
#define HC    64   // HEADS*CH
#define KDIM  128

// -------------------------------------------------------------------------
// TDM: DMA W (128x64 f32 = 32 KB) from global into LDS as one 2D tile.
// D# built per CDNA5 ISA 8.3/8.4: count=1, type=2(image), data_size=4B,
// tensor_dim0 = tile_dim0 = 8192 elements, tensor_dim1 = tile_dim1 = 1,
// tensor_dim0_stride = 8192. Tracked by TENSORcnt.
// -------------------------------------------------------------------------
#if __has_builtin(__builtin_amdgcn_tensor_load_to_lds)
#define HAVE_TDM 1
__device__ __forceinline__ void tdm_load_W(const float* W, unsigned lds_off)
{
    unsigned long long ga = (unsigned long long)(uintptr_t)W;
    u32x4 g0;
    g0[0] = 1u;                                              // count=1, no gather
    g0[1] = lds_off;                                         // lds_addr (bits 63:32)
    g0[2] = (unsigned)(ga & 0xFFFFFFFFu);                    // global_addr lo
    g0[3] = (unsigned)((ga >> 32) & 0x01FFFFFFu)             // global_addr hi (57b)
          | 0x80000000u;                                     // type=2 (bits 127:126)
    i32x8 g1 = {};
    g1[0] = 2 << 16;                                         // data_size=2 -> 4 bytes
    g1[1] = (int)(0x2000u << 16);                            // tensor_dim0=8192 (63:48)
    g1[2] = (int)(1u << 16);                                 // tensor_dim1=1   (95:80)
    g1[3] = (int)(0x2000u << 16);                            // tile_dim0=8192  (127:112)
    g1[4] = 1;                                               // tile_dim1=1     (143:128)
    g1[5] = 8192;                                            // dim0_stride     (191:160)
    i32x4 z4 = {};
#if defined(__clang_major__) && (__clang_major__ >= 23)
    i32x8 z8 = {};
    __builtin_amdgcn_tensor_load_to_lds(g0, g1, z4, z4, z8, 0);
#else
    __builtin_amdgcn_tensor_load_to_lds(g0, g1, z4, z4, 0);
#endif
}
#else
#define HAVE_TDM 0
#endif

// -------------------------------------------------------------------------
// Kernel 1: h = x @ W   (M x 128) @ (128 x 64), f16 WMMA with f32 accumulate
// Block = 128 threads = 4 wave32; block handles 16 rows, each wave one
// 16-column tile. W is DMA'd to LDS (f32) by the TDM, then pre-swizzled
// into per-lane contiguous B fragments (f16) so the inner loop is
// 2 x ds_load_b128 per WMMA instead of 16 x ds_load_u16.
// -------------------------------------------------------------------------
__global__ void gat_gemm_wmma(const float* __restrict__ x,
                              const float* __restrict__ W,
                              float* __restrict__ h, int M)
{
    __shared__ float Wf[KDIM * HC];                               // 32 KB
    __shared__ _Float16 Bfrag[512 * 16] __attribute__((aligned(32))); // 16 KB

    const int tid  = threadIdx.x;
    const int lane = tid & 31;
    const int wave = tid >> 5;           // 0..3 -> column tile
    const int n0   = wave * 16;
    const int m0   = blockIdx.x * 16;
    const int hi   = (lane < 16) ? 0 : 1;
    const int l15  = lane & 15;

    // ---- stage W in LDS (f32) via Tensor Data Mover ----
#if HAVE_TDM
    if (tid < 32) {
        tdm_load_W(W, (unsigned)(uintptr_t)&Wf[0]);
        __builtin_amdgcn_s_wait_tensorcnt(0);
    }
    __syncthreads();
#else
    for (int i = tid; i < KDIM * HC; i += blockDim.x) Wf[i] = W[i];
    __syncthreads();
#endif

    // ---- pre-swizzle into WMMA B-fragment order (f16) ----
    // fragment f = ((w*4 + kbi)*32 + lane); lane's 16 elements contiguous.
    for (int f = tid; f < 512; f += blockDim.x) {
        const int fl    = f & 31;
        const int kbi   = (f >> 5) & 3;
        const int w     = f >> 7;
        const int kbase = kbi * 32 + ((fl < 16) ? 0 : 8);
        const int col   = w * 16 + (fl & 15);
        _Float16* dst = &Bfrag[f * 16];
        #pragma unroll
        for (int i = 0; i < 16; ++i)
            dst[i] = (_Float16)Wf[(kbase + i + ((i >= 8) ? 8 : 0)) * HC + col];
    }
    __syncthreads();

    int arow = m0 + l15;
    if (arow >= M) arow = M - 1;         // clamp (M is a multiple of 16 anyway)
    const float4* xr = (const float4*)(x + (size_t)arow * KDIM);

    v8f acc = {};
    #pragma unroll
    for (int kb = 0; kb < KDIM; kb += 32) {
        // ---- A fragment: 16-bit A 16x32 layout ----
        // lanes 0-15 : K = kb+0..7   and kb+16..23
        // lanes 16-31: K = kb+8..15  and kb+24..31
        const int koff = hi ? 8 : 0;
        float4 q0 = xr[(kb + koff) >> 2];
        float4 q1 = xr[((kb + koff) >> 2) + 1];
        float4 q2 = xr[(kb + 16 + koff) >> 2];
        float4 q3 = xr[((kb + 16 + koff) >> 2) + 1];
        v16h a;
        a[0]=(_Float16)q0.x; a[1]=(_Float16)q0.y; a[2]=(_Float16)q0.z; a[3]=(_Float16)q0.w;
        a[4]=(_Float16)q1.x; a[5]=(_Float16)q1.y; a[6]=(_Float16)q1.z; a[7]=(_Float16)q1.w;
        a[8]=(_Float16)q2.x; a[9]=(_Float16)q2.y; a[10]=(_Float16)q2.z; a[11]=(_Float16)q2.w;
        a[12]=(_Float16)q3.x; a[13]=(_Float16)q3.y; a[14]=(_Float16)q3.z; a[15]=(_Float16)q3.w;

        // ---- B fragment: one contiguous 32B LDS chunk per lane ----
        const int kbi = kb >> 5;
        const v16h b = *(const v16h*)(&Bfrag[(((wave << 2) + kbi) * 32 + lane) * 16]);

        acc = __builtin_amdgcn_wmma_f32_16x16x32_f16(
                  false, a, false, b, (short)0, acc, false, false);
    }

    // ---- C store: f32 C/D layout (VGPR v -> M = v + 8*hi, N = l15) ----
    #pragma unroll
    for (int v = 0; v < 8; ++v) {
        int row = m0 + v + (hi ? 8 : 0);
        if (row < M)
            h[(size_t)row * HC + n0 + l15] = acc[v];
    }
}

// -------------------------------------------------------------------------
// Kernel 2: per-(node,head) attention coefficients
// -------------------------------------------------------------------------
__global__ void gat_attn_coef(const float* __restrict__ h,
                              const float* __restrict__ att_src,
                              const float* __restrict__ att_dst,
                              float* __restrict__ a_src,
                              float* __restrict__ a_dst, int Nn)
{
    int t = blockIdx.x * blockDim.x + threadIdx.x;
    if (t >= Nn * HEADS) return;
    int n = t / HEADS, hh = t % HEADS;
    const float* hp = h + (size_t)n * HC + hh * CH;
    const float* as = att_src + hh * CH;
    const float* ad = att_dst + hh * CH;
    float s = 0.f, d = 0.f;
    #pragma unroll
    for (int c = 0; c < CH; ++c) { float v = hp[c]; s += v * as[c]; d += v * ad[c]; }
    a_src[t] = s;
    a_dst[t] = d;
}

// -------------------------------------------------------------------------
// Kernel 3: init accumulators (m = -inf, denom = 0, out = 0)
// -------------------------------------------------------------------------
__global__ void gat_init(float* __restrict__ m, float* __restrict__ denom,
                         float* __restrict__ out, int Nn)
{
    int t = blockIdx.x * blockDim.x + threadIdx.x;
    if (t < Nn * HEADS) { m[t] = -__builtin_inff(); denom[t] = 0.f; }
    if (t < Nn * HC) out[t] = 0.f;
}

__device__ __forceinline__ void atomicMaxF(float* a, float v)
{
    if (v >= 0.f) atomicMax((int*)a, __float_as_int(v));
    else          atomicMin((unsigned int*)a, __float_as_uint(v));
}

__device__ __forceinline__ void edge_nodes(const int* __restrict__ ei, int E,
                                           int e, int& src, int& dst)
{
    if (e < E) { src = ei[e]; dst = ei[E + e]; }
    else       { src = dst = e - E; }        // self-loop
}

// -------------------------------------------------------------------------
// Kernel 4: segment max over destination (leaky-relu logits)
// -------------------------------------------------------------------------
__global__ void gat_edge_max(const int* __restrict__ ei, int E, int Etot,
                             const float* __restrict__ a_src,
                             const float* __restrict__ a_dst,
                             float* __restrict__ m)
{
    int e = blockIdx.x * blockDim.x + threadIdx.x;
    if (e >= Etot) return;
    int src, dst; edge_nodes(ei, E, e, src, dst);
    #pragma unroll
    for (int hh = 0; hh < HEADS; ++hh) {
        float v = a_src[src * HEADS + hh] + a_dst[dst * HEADS + hh];
        v = v > 0.f ? v : 0.2f * v;
        atomicMaxF(&m[dst * HEADS + hh], v);
    }
}

// -------------------------------------------------------------------------
// Kernel 5: replace non-finite maxima with 0 (matches reference)
// -------------------------------------------------------------------------
__global__ void gat_fix_m(float* __restrict__ m, int Nn)
{
    int t = blockIdx.x * blockDim.x + threadIdx.x;
    if (t >= Nn * HEADS) return;
    float v = m[t];
    if (!__builtin_isfinite(v)) m[t] = 0.f;
}

// -------------------------------------------------------------------------
// Kernel 6: segment sum of exp(e - m[dst])
// -------------------------------------------------------------------------
__global__ void gat_edge_sum(const int* __restrict__ ei, int E, int Etot,
                             const float* __restrict__ a_src,
                             const float* __restrict__ a_dst,
                             const float* __restrict__ m,
                             float* __restrict__ denom)
{
    int e = blockIdx.x * blockDim.x + threadIdx.x;
    if (e >= Etot) return;
    int src, dst; edge_nodes(ei, E, e, src, dst);
    #pragma unroll
    for (int hh = 0; hh < HEADS; ++hh) {
        float v = a_src[src * HEADS + hh] + a_dst[dst * HEADS + hh];
        v = v > 0.f ? v : 0.2f * v;
        float ex = __expf(v - m[dst * HEADS + hh]);
        atomicAdd(&denom[dst * HEADS + hh], ex);
    }
}

// -------------------------------------------------------------------------
// Kernel 7: scatter alpha * h[src] into out[dst]; thread per (edge, head)
// -------------------------------------------------------------------------
__global__ void gat_edge_aggr(const int* __restrict__ ei, int E, int Etot,
                              const float* __restrict__ a_src,
                              const float* __restrict__ a_dst,
                              const float* __restrict__ m,
                              const float* __restrict__ denom,
                              const float* __restrict__ h,
                              float* __restrict__ out)
{
    int t = blockIdx.x * blockDim.x + threadIdx.x;
    if (t >= Etot * HEADS) return;
    int e  = t >> 3;
    int hh = t & 7;
    int src, dst; edge_nodes(ei, E, e, src, dst);
    float v = a_src[src * HEADS + hh] + a_dst[dst * HEADS + hh];
    v = v > 0.f ? v : 0.2f * v;
    float ex    = __expf(v - m[dst * HEADS + hh]);
    float alpha = ex / (denom[dst * HEADS + hh] + 1e-16f);
    const float* hp = h   + (size_t)src * HC + hh * CH;
    float*       op = out + (size_t)dst * HC + hh * CH;
    #pragma unroll
    for (int c = 0; c < CH; ++c)
        atomicAdd(op + c, alpha * hp[c]);
}

// -------------------------------------------------------------------------
// Kernel 8: out = relu(out + bias)
// -------------------------------------------------------------------------
__global__ void gat_finalize(float* __restrict__ out,
                             const float* __restrict__ bias, int Nn)
{
    int t = blockIdx.x * blockDim.x + threadIdx.x;
    if (t >= Nn * HC) return;
    float v = out[t] + bias[t & (HC - 1)];
    out[t] = v > 0.f ? v : 0.f;
}

// -------------------------------------------------------------------------
extern "C" void kernel_launch(void* const* d_in, const int* in_sizes, int n_in,
                              void* d_out, int out_size, void* d_ws, size_t ws_size,
                              hipStream_t stream)
{
    (void)n_in; (void)out_size; (void)ws_size;
    const float* x       = (const float*)d_in[0];
    const int*   ei      = (const int*)  d_in[1];
    const float* W       = (const float*)d_in[2];
    const float* att_src = (const float*)d_in[3];
    const float* att_dst = (const float*)d_in[4];
    const float* bias    = (const float*)d_in[5];
    float*       out     = (float*)d_out;

    const int Nn   = in_sizes[0] / KDIM;   // 100000
    const int E    = in_sizes[1] / 2;      // 1600000
    const int Etot = E + Nn;               // with self-loops

    // workspace layout (f32): h[Nn*64] | a_src[Nn*8] | a_dst[Nn*8] | m[Nn*8] | denom[Nn*8]
    float* ws    = (float*)d_ws;
    float* h     = ws;
    float* a_src = h     + (size_t)Nn * HC;
    float* a_dst = a_src + (size_t)Nn * HEADS;
    float* m     = a_dst + (size_t)Nn * HEADS;
    float* denom = m     + (size_t)Nn * HEADS;

    gat_gemm_wmma<<<(Nn + 15) / 16, 128, 0, stream>>>(x, W, h, Nn);
    gat_init<<<(Nn * HC + 255) / 256, 256, 0, stream>>>(m, denom, out, Nn);
    gat_attn_coef<<<(Nn * HEADS + 255) / 256, 256, 0, stream>>>(h, att_src, att_dst,
                                                                a_src, a_dst, Nn);
    gat_edge_max<<<(Etot + 255) / 256, 256, 0, stream>>>(ei, E, Etot, a_src, a_dst, m);
    gat_fix_m<<<(Nn * HEADS + 255) / 256, 256, 0, stream>>>(m, Nn);
    gat_edge_sum<<<(Etot + 255) / 256, 256, 0, stream>>>(ei, E, Etot, a_src, a_dst,
                                                         m, denom);
    gat_edge_aggr<<<(Etot * HEADS + 255) / 256, 256, 0, stream>>>(ei, E, Etot, a_src,
                                                                  a_dst, m, denom, h, out);
    gat_finalize<<<(Nn * HC + 255) / 256, 256, 0, stream>>>(out, bias, Nn);
}
